// DiscriminationLoss_32908039422364
// MI455X (gfx1250) — compile-verified, matrix operand order checked by reading, not run on hardware
//
#include <hip/hip_runtime.h>
#include <math.h>

// ---------------------------------------------------------------------------
// DiscriminationLoss on MI455X (gfx1250, wave32, WMMA).
//
// Pass 1 (segsum_wmma_kernel): streaming segmented sum via
//   V_WMMA_F32_16X16X32_F16:  D[k][n] += onehot(lab==k) * pred[p][n]
//   rows(M)=16 label bins (9 used), cols(N)=8 channels + ones column (counts),
//   K=32 pixels per WMMA, 2 WMMAs per 64-pixel chunk.
//   Memory-bound: 118 MB -> ~5 us floor @ 23.3 TB/s; b64 coalesced loads,
//   v_cvt_pk_rtz_f16_f32 packing, ds_bpermute one-hot build keep VALU low.
// Pass 2 (finalize_kernel): 648-float epilogue -> scalar loss.
// ---------------------------------------------------------------------------

typedef __attribute__((ext_vector_type(16))) _Float16 v16h;
typedef __attribute__((ext_vector_type(8)))  float    v8f;
typedef __attribute__((ext_vector_type(2)))  __fp16   fp16x2;  // cvt_pkrtz result type

#define KBINS   9                 // labels 0..8
#define CCH     8                 // channels
#define PPIX    (640 * 640)       // pixels per image
#define NBATCH  8
#define WS_FLOATS (NBATCH * KBINS * (CCH + 1))   // s[b][k][0..7] + count at [8] = 648

#define ROWB    20                // LDS bytes per staged pixel row (16B f16 + 4B pad)
#define CHUNK   64                // pixels per wave iteration (2 WMMAs)
#define SLAB    1344              // per-wave slab bytes (64*20 = 1280 + overread slack)

__global__ void zero_ws_kernel(float* __restrict__ ws) {
    int i = blockIdx.x * blockDim.x + threadIdx.x;
    if (i < WS_FLOATS) ws[i] = 0.0f;
}

__device__ __forceinline__ unsigned pack2h(float a, float b) {
    union { fp16x2 h; unsigned u; } u;
    u.h = __builtin_amdgcn_cvt_pkrtz(a, b);      // v_cvt_pk_rtz_f16_f32, 1 VALU op
    return u.u;
}

union V16H { unsigned u[8]; v16h v; };

__global__ void __launch_bounds__(256)
segsum_wmma_kernel(const float* __restrict__ pred,
                   const int*   __restrict__ labels,
                   float*       __restrict__ ws)
{
    __shared__ unsigned stage_u[8 * (SLAB / 4)];   // per-wave f16 staging slabs
    __shared__ float    red[8 * 256];              // block reduction of accumulators

    const int b       = blockIdx.y;
    const int lane    = threadIdx.x & 31;
    const int wave    = threadIdx.x >> 5;
    const int waveG   = blockIdx.x * 8 + wave;
    const int wstride = gridDim.x * 8;

    const float* predB = pred   + (size_t)b * CCH * PPIX;
    const int2*  labB2 = (const int2*)(labels + (size_t)b * PPIX);

    unsigned* slab_u = stage_u + wave * (SLAB / 4);
    const unsigned char* slab_c = (const unsigned char*)slab_u;

    // Per-lane constants for the ISA 16-bit WMMA fragment layouts:
    //  A (16x32 one-hot): lane row m = lane&15; K-slot pair for dword j is
    //    k0 = 2j + 8*(j>=4) + 8*(lane>=16), k1 = k0+1 (k0 always even).
    //  B (32x16 pred):    lane column n = lane&15; same K mapping.
    const int  m        = lane & 15;
    const int  lane8    = (lane >> 4) << 3;          // 0 or 8
    const int  n2       = (lane & 15) * 2;           // byte offset of column in row
    const bool isChan   = (lane & 15) < CCH;
    const unsigned constSel = ((lane & 15) == CCH) ? 0x3C003C00u : 0u; // ones col / zero

    v8f acc = {};

    const int nchunks = PPIX / CHUNK;                // 6400, exact
    for (int chunk = waveG; chunk < nchunks; chunk += wstride) {
        const int ppair = chunk * (CHUNK / 2) + lane;   // index into float2/int2 views

        // --- coalesced 256B b64 loads: labels + 8 channel planes, 2 px/lane ---
        const int2 ll = labB2[ppair];
        float2 x[CCH];
        #pragma unroll
        for (int c = 0; c < CCH; ++c)
            x[c] = ((const float2*)(predB + c * PPIX))[ppair];

        // --- stage as f16 [pixel][channel] rows (20B padded stride) ---
        // lane owns pixel rows 2*lane and 2*lane+1; 10*lane+c2 distinct mod 64
        unsigned* row0 = slab_u + (2 * lane) * 5;
        row0[0] = pack2h(x[0].x, x[1].x);
        row0[1] = pack2h(x[2].x, x[3].x);
        row0[2] = pack2h(x[4].x, x[5].x);
        row0[3] = pack2h(x[6].x, x[7].x);
        unsigned* row1 = row0 + 5;
        row1[0] = pack2h(x[0].y, x[1].y);
        row1[1] = pack2h(x[2].y, x[3].y);
        row1[2] = pack2h(x[4].y, x[5].y);
        row1[3] = pack2h(x[6].y, x[7].y);

        // --- two 16x16x32 WMMAs over the 64-pixel chunk ---
        #pragma unroll
        for (int w = 0; w < 2; ++w) {
            V16H A, Bm;
            #pragma unroll
            for (int j = 0; j < 8; ++j) {
                const int kbase = 2 * j + ((j >= 4) ? 8 : 0);
                const int klane = kbase + lane8;          // even, 0..30

                // labels of pixels (32w+klane, 32w+klane+1) live in lane
                // 16w + klane/2 as (ll.x, ll.y): one shared bpermute index.
                const int idx = w * 64 + klane * 2;       // bytes
                const int l0  = __builtin_amdgcn_ds_bpermute(idx, ll.x);
                const int l1  = __builtin_amdgcn_ds_bpermute(idx, ll.y);
                A.u[j] = ((l0 == m) ? 0x3C00u : 0u) | ((l1 == m) ? 0x3C000000u : 0u);

                const int off = (w * 32 + klane) * ROWB + n2;
                const unsigned short b0 = *(const unsigned short*)(slab_c + off);
                const unsigned short b1 = *(const unsigned short*)(slab_c + off + ROWB);
                unsigned bp = (unsigned)b0 | ((unsigned)b1 << 16);
                Bm.u[j] = isChan ? bp : constSel;
            }
            // D[k][n] += sum_p onehot(lab[p]==k) * pred[p][n]
            acc = __builtin_amdgcn_wmma_f32_16x16x32_f16(
                      false, A.v, false, Bm.v, (short)0, acc, false, false);
        }
    }

    // --- block reduction (8 waves -> wave 0), then 81 global f32 atomics ---
    #pragma unroll
    for (int r = 0; r < 8; ++r) red[wave * 256 + r * 32 + lane] = acc[r];
    __syncthreads();

    if (wave == 0) {
        #pragma unroll
        for (int r = 0; r < 8; ++r) {
            float s = 0.0f;
            #pragma unroll
            for (int w = 0; w < 8; ++w) s += red[w * 256 + r * 32 + lane];
            // C/D layout: lanes 0-15 -> M=r, lanes 16-31 -> M=8+r; N = lane&15
            const int kk = r + lane8;
            const int nn = lane & 15;
            if (kk < KBINS && nn <= CCH)
                atomicAdd(&ws[(b * KBINS + kk) * (CCH + 1) + nn], s);
        }
    }
}

__global__ void finalize_kernel(const float* __restrict__ ws, float* __restrict__ out) {
    __shared__ float partial[NBATCH];
    const int t = threadIdx.x;
    const float f_sigma = logf(10.0f);               // log(3^2 + 1)

    if (t < NBATCH) {
        const float* wb = ws + t * KBINS * (CCH + 1);
        float sum_term = 0.0f;
        int   mmax     = 0;
        for (int k = 1; k <= 8; ++k) {
            const float* row = wb + k * (CCH + 1);
            float nrm2 = 0.0f;
            #pragma unroll
            for (int c = 0; c < CCH; ++c) { float v = row[c]; nrm2 += v * v; }
            const float snorm = sqrtf(nrm2);
            const float r     = fmaxf(3.0f - snorm, 0.0f);
            const float f     = logf(r * r + 1.0f);
            const float n_k   = row[CCH];
            sum_term += n_k * (f - f_sigma);
            if (n_k > 0.0f) mmax = k;                // == lab.max() for this batch
        }
        const float mf    = (float)mmax;
        const float pairs = mf * (mf - 1.0f) * 0.5f;
        const float BP    = (float)NBATCH * (float)PPIX;
        partial[t] = (mf - 1.0f) * sum_term + pairs * BP * f_sigma;
    }
    __syncthreads();
    if (t == 0) {
        float s = 0.0f;
        #pragma unroll
        for (int i = 0; i < NBATCH; ++i) s += partial[i];
        out[0] = s;
    }
}

extern "C" void kernel_launch(void* const* d_in, const int* in_sizes, int n_in,
                              void* d_out, int out_size, void* d_ws, size_t ws_size,
                              hipStream_t stream) {
    (void)in_sizes; (void)n_in; (void)out_size; (void)ws_size;
    const float* pred   = (const float*)d_in[0];   // [8,8,640,640] f32
    const int*   labels = (const int*)d_in[1];     // [8,1,640,640] i32
    float* out = (float*)d_out;
    float* ws  = (float*)d_ws;                     // 648 floats of scratch

    zero_ws_kernel<<<(WS_FLOATS + 255) / 256, 256, 0, stream>>>(ws);

    dim3 grid(128, NBATCH);                        // 1024 blocks x 256 threads (8 waves)
    segsum_wmma_kernel<<<grid, 256, 0, stream>>>(pred, labels, ws);

    finalize_kernel<<<1, 32, 0, stream>>>(ws, out);
}
